// MultiModalAttention_14267881357602
// MI455X (gfx1250) — matrix operand order repaired and checked by reading, not compiled, not run on hardware
//
#include <hip/hip_runtime.h>
#include <stdint.h>

// ---------------------------------------------------------------------------
// MultiModalAttention for MI455X (gfx1250, wave32, WMMA 16x16x32 f16,
// async global->LDS tile staging with double buffering)
// ---------------------------------------------------------------------------

typedef __attribute__((ext_vector_type(16))) _Float16 v16h;
typedef __attribute__((ext_vector_type(8)))  _Float16 v8h;
typedef __attribute__((ext_vector_type(8)))  float    v8f;

#define WMMA_F32_F16(a, b, c) \
  __builtin_amdgcn_wmma_f32_16x16x32_f16(false, (a), false, (b), (short)0, (c), false, false)

// Async global->LDS 16B copy (per lane). LDS offset = low 32 bits of the flat
// address (ISA 10.2: LDS aperture truncates to addr[31:0]). Tracked by ASYNCcnt.
#define ASYNC_LDS_B128(ldsoff, gaddr) \
  asm volatile("global_load_async_to_lds_b128 %0, %1, off" :: "v"(ldsoff), "v"(gaddr) : "memory")
#define WAIT_ASYNC0() asm volatile("s_wait_asynccnt 0x0" ::: "memory")

__device__ __forceinline__ uint32_t lds_off(const void* p) {
  return (uint32_t)(uintptr_t)p;
}

static constexpr int S_DIM = 1024;
static constexpr int D_DIM = 1024;
static constexpr int H_NUM = 16;
static constexpr int DK_DIM = 64;

// A fragment (16x32 f16): lane L -> row L%16, k in [8*hi,8*hi+8) u [16+8*hi,+8)
__device__ __forceinline__ v16h frag_a(const _Float16* rowp, int hi) {
  v8h lo = *(const v8h*)(rowp + 8 * hi);
  v8h h8 = *(const v8h*)(rowp + 16 + 8 * hi);
  return __builtin_shufflevector(lo, h8, 0, 1, 2, 3, 4, 5, 6, 7, 8, 9, 10, 11, 12, 13, 14, 15);
}

// B fragment (32x16 f16): lane L -> col L%16, k in [8*hi,8*hi+8) u [16+8*hi,+8)
__device__ __forceinline__ v16h frag_b(const _Float16* rowp, int hi) {
  v8h lo = *(const v8h*)(rowp + 8 * hi);
  v8h h8 = *(const v8h*)(rowp + 16 + 8 * hi);
  return __builtin_shufflevector(lo, h8, 0, 1, 2, 3, 4, 5, 6, 7, 8, 9, 10, 11, 12, 13, 14, 15);
}

// ---------------------------------------------------------------------------
// f32 -> f16 conversion, 8 elements / thread
// ---------------------------------------------------------------------------
__global__ void cvt_f32_f16_k(const float* __restrict__ in, _Float16* __restrict__ out, int n) {
  int i = (blockIdx.x * 256 + threadIdx.x) * 8;
  if (i >= n) return;
  float4 a = *(const float4*)(in + i);
  float4 b = *(const float4*)(in + i + 4);
  v8h r;
  r[0] = (_Float16)a.x; r[1] = (_Float16)a.y; r[2] = (_Float16)a.z; r[3] = (_Float16)a.w;
  r[4] = (_Float16)b.x; r[5] = (_Float16)b.y; r[6] = (_Float16)b.z; r[7] = (_Float16)b.w;
  *(v8h*)(out + i) = r;
}

// ---------------------------------------------------------------------------
// WMMA GEMM: Out = A[M,K] * Bw[N,K]^T + bias, fp32 accumulate.
// Double-buffered LDS, async global->LDS staging.
// store_mode: 0 = head-layout f16 ([B,H,S,64]), 1 = row-major f16, 2 = row-major f32
// Block: 256 threads (8 waves), tile 128x64, K-step 32.
// ---------------------------------------------------------------------------
enum { ST_HEAD_F16 = 0, ST_ROW_F16 = 1, ST_ROW_F32 = 2 };

__global__ __launch_bounds__(256) void gemm_wmma_k(
    const _Float16* __restrict__ A, const _Float16* __restrict__ Bw,
    const float* __restrict__ bias, void* __restrict__ Out,
    int M, int N, int K, int store_mode, int do_relu) {
  __shared__ __align__(16) _Float16 As[2][128][40];  // 32-wide k slice + 8 pad
  __shared__ __align__(16) _Float16 Bs[2][64][40];
  const int t = threadIdx.x;
  const int lane = t & 31, wid = t >> 5;
  const int hi = lane >> 4, li = lane & 15;
  const int wm = wid >> 1, wn = wid & 1;  // 4 x 2 wave grid
  const int m0 = blockIdx.x * 128, n0 = blockIdx.y * 64;

  const int r4 = t >> 2, o4 = (t & 3) << 3;  // 16B chunk coords (4 chunks/row)

  auto stage = [&](int k0, int buf) {
    // A tile: 128 rows x 32 halves = 512 x 16B chunks, 2 per thread
    ASYNC_LDS_B128(lds_off(&As[buf][r4][o4]),
                   (uint64_t)(uintptr_t)(A + (size_t)(m0 + r4) * K + k0 + o4));
    ASYNC_LDS_B128(lds_off(&As[buf][r4 + 64][o4]),
                   (uint64_t)(uintptr_t)(A + (size_t)(m0 + r4 + 64) * K + k0 + o4));
    // B tile: 64 rows x 32 halves = 256 x 16B chunks, 1 per thread
    ASYNC_LDS_B128(lds_off(&Bs[buf][r4][o4]),
                   (uint64_t)(uintptr_t)(Bw + (size_t)(n0 + r4) * K + k0 + o4));
  };

  v8f acc[2][2] = {};

  stage(0, 0);
  WAIT_ASYNC0();
  __syncthreads();

  const int KT = K >> 5;
  for (int kt = 0; kt < KT; ++kt) {
    const int cur = kt & 1;
    if (kt + 1 < KT) stage((kt + 1) << 5, cur ^ 1);  // prefetch next tile

    #pragma unroll
    for (int tm = 0; tm < 2; ++tm) {
      v16h a = frag_a(&As[cur][wm * 32 + tm * 16 + li][0], hi);
      #pragma unroll
      for (int tn = 0; tn < 2; ++tn) {
        v16h b = frag_b(&Bs[cur][wn * 32 + tn * 16 + li][0], hi);
        acc[tm][tn] = WMMA_F32_F16(a, b, acc[tm][tn]);
      }
    }
    WAIT_ASYNC0();
    __syncthreads();
  }

  // epilogue: bias (+relu) + store. C layout: lane -> col li, rows v + 8*hi.
  #pragma unroll
  for (int tm = 0; tm < 2; ++tm) {
    #pragma unroll
    for (int tn = 0; tn < 2; ++tn) {
      int col = n0 + wn * 32 + tn * 16 + li;
      float bv = bias ? bias[col] : 0.0f;
      #pragma unroll
      for (int v = 0; v < 8; ++v) {
        int row = m0 + wm * 32 + tm * 16 + v + 8 * hi;
        float x = acc[tm][tn][v] + bv;
        if (do_relu) x = fmaxf(x, 0.0f);
        if (store_mode == ST_HEAD_F16) {
          int bidx = row / S_DIM, s = row % S_DIM;
          int h = col >> 6, dk = col & 63;
          ((_Float16*)Out)[((((size_t)bidx * H_NUM + h) * S_DIM + s) << 6) + dk] = (_Float16)x;
        } else if (store_mode == ST_ROW_F16) {
          ((_Float16*)Out)[(size_t)row * N + col] = (_Float16)x;
        } else {
          ((float*)Out)[(size_t)row * N + col] = x;
        }
      }
    }
  }
}

// ---------------------------------------------------------------------------
// Flash attention: heads layout [B,H,S,64] f16 for Q/K/V.
// Block: 256 threads (8 waves) handles 128 q rows of one (b,h); wave owns 16 rows.
// Double-buffered K/V tiles (K async->LDS, V transposed through registers),
// online softmax with 16-lane butterfly reductions, P bounced through LDS.
// out_is_f32: 0 -> f16 head-layout write; 1 -> f32 head-layout (init or +=).
// ---------------------------------------------------------------------------
__global__ __launch_bounds__(256) void attn_wmma_k(
    const _Float16* __restrict__ Qh, const _Float16* __restrict__ Kh,
    const _Float16* __restrict__ Vh, const int* __restrict__ maskp,
    void* __restrict__ Out, int out_is_f32, int do_init) {
  __shared__ __align__(16) _Float16 Ks[2][32][72];     // 32 k-cols x 64 dk (+8 pad)
  __shared__ __align__(16) _Float16 Vt[2][64][40];     // V transposed: [dk][k] (+8 pad)
  __shared__ __align__(16) _Float16 Pl[8][16][40];     // per-wave P tile 16x32 (+8 pad)

  const int t = threadIdx.x, lane = t & 31, wid = t >> 5;
  const int hi = lane >> 4, li = lane & 15;
  const int bh = blockIdx.y, b = bh / H_NUM;
  const int q0 = blockIdx.x * 128 + wid * 16;

  const _Float16* qrow = Qh + (((size_t)bh * S_DIM + q0 + li) << 6);
  v16h aq0 = frag_a(qrow, hi);
  v16h aq1 = frag_a(qrow + 32, hi);

  const _Float16* Kb = Kh + (((size_t)bh * S_DIM) << 6);
  const _Float16* Vb = Vh + (((size_t)bh * S_DIM) << 6);

  const int r8 = t >> 3, c8 = (t & 7) << 3;  // 32 rows x 8 chunks of 16B

  auto stageKV = [&](int kc, int buf) {
    ASYNC_LDS_B128(lds_off(&Ks[buf][r8][c8]),
                   (uint64_t)(uintptr_t)(Kb + (((size_t)(kc + r8)) << 6) + c8));
    v8h vv = *(const v8h*)(Vb + (((size_t)(kc + r8)) << 6) + c8);
    #pragma unroll
    for (int j = 0; j < 8; ++j) Vt[buf][c8 + j][r8] = vv[j];
  };

  v8f o0 = {}, o1 = {}, o2 = {}, o3 = {};
  float mrow[8], lrow[8];
  #pragma unroll
  for (int v = 0; v < 8; ++v) { mrow[v] = -1e30f; lrow[v] = 0.0f; }

  stageKV(0, 0);
  WAIT_ASYNC0();
  __syncthreads();

  for (int kc = 0; kc < S_DIM; kc += 32) {
    const int cur = (kc >> 5) & 1;
    if (kc + 32 < S_DIM) stageKV(kc + 32, cur ^ 1);  // prefetch next tile

    // S tiles: cols [kc, kc+16) and [kc+16, kc+32), K-dim 64 = 2 WMMA steps
    v8f s0 = {}, s1 = {};
    s0 = WMMA_F32_F16(aq0, frag_b(&Ks[cur][li][0], hi), s0);
    s0 = WMMA_F32_F16(aq1, frag_b(&Ks[cur][li][32], hi), s0);
    s1 = WMMA_F32_F16(aq0, frag_b(&Ks[cur][16 + li][0], hi), s1);
    s1 = WMMA_F32_F16(aq1, frag_b(&Ks[cur][16 + li][32], hi), s1);

    bool ok0 = true, ok1 = true;
    if (maskp) {
      ok0 = maskp[(size_t)b * S_DIM + kc + li] != 0;
      ok1 = maskp[(size_t)b * S_DIM + kc + 16 + li] != 0;
    }

    #pragma unroll
    for (int v = 0; v < 8; ++v) {
      float a0 = ok0 ? s0[v] * 0.125f : -1e9f;  // 1/sqrt(64)
      float a1 = ok1 ? s1[v] * 0.125f : -1e9f;
      float tmx = fmaxf(a0, a1);
      tmx = fmaxf(tmx, __shfl_xor(tmx, 1, 32));
      tmx = fmaxf(tmx, __shfl_xor(tmx, 2, 32));
      tmx = fmaxf(tmx, __shfl_xor(tmx, 4, 32));
      tmx = fmaxf(tmx, __shfl_xor(tmx, 8, 32));
      float mn = fmaxf(mrow[v], tmx);
      float alpha = __expf(mrow[v] - mn);
      mrow[v] = mn;
      float p0 = __expf(a0 - mn);
      float p1 = __expf(a1 - mn);
      float ps = p0 + p1;
      ps += __shfl_xor(ps, 1, 32);
      ps += __shfl_xor(ps, 2, 32);
      ps += __shfl_xor(ps, 4, 32);
      ps += __shfl_xor(ps, 8, 32);
      lrow[v] = lrow[v] * alpha + ps;
      o0[v] *= alpha; o1[v] *= alpha; o2[v] *= alpha; o3[v] *= alpha;
      // C layout -> P[m][k]: m = v + 8*hi, k = (tile)*16 + li
      Pl[wid][v + 8 * hi][li] = (_Float16)p0;
      Pl[wid][v + 8 * hi][16 + li] = (_Float16)p1;
    }

    // P*V: P is the A operand (re-fragment from per-wave LDS; ds ops in-order per wave)
    v16h ap = frag_a(&Pl[wid][li][0], hi);
    o0 = WMMA_F32_F16(ap, frag_b(&Vt[cur][li][0], hi), o0);
    o1 = WMMA_F32_F16(ap, frag_b(&Vt[cur][16 + li][0], hi), o1);
    o2 = WMMA_F32_F16(ap, frag_b(&Vt[cur][32 + li][0], hi), o2);
    o3 = WMMA_F32_F16(ap, frag_b(&Vt[cur][48 + li][0], hi), o3);

    WAIT_ASYNC0();
    __syncthreads();
  }

  // epilogue: divide by l, store head-layout
  #pragma unroll
  for (int v = 0; v < 8; ++v) {
    float inv = 1.0f / lrow[v];
    int row = q0 + v + 8 * hi;
    size_t base = ((size_t)bh * S_DIM + row) << 6;
    float r0 = o0[v] * inv, r1 = o1[v] * inv, r2 = o2[v] * inv, r3 = o3[v] * inv;
    if (out_is_f32) {
      float* O = (float*)Out;
      if (do_init) {
        O[base + li] = r0; O[base + 16 + li] = r1;
        O[base + 32 + li] = r2; O[base + 48 + li] = r3;
      } else {
        O[base + li] += r0; O[base + 16 + li] += r1;
        O[base + 32 + li] += r2; O[base + 48 + li] += r3;
      }
    } else {
      _Float16* O = (_Float16*)Out;
      O[base + li] = (_Float16)r0; O[base + 16 + li] = (_Float16)r1;
      O[base + 32 + li] = (_Float16)r2; O[base + 48 + li] = (_Float16)r3;
    }
  }
}

// ---------------------------------------------------------------------------
// combine: conc[b,s,h*64+dk] = E + image_bool * I  (head -> concat layout, f16)
// ---------------------------------------------------------------------------
__global__ void combine_k(const float* __restrict__ E, const float* __restrict__ I,
                          const int* __restrict__ img, _Float16* __restrict__ conc, int total) {
  int i = blockIdx.x * 256 + threadIdx.x;
  if (i >= total) return;
  int dk = i & 63;
  int r = i >> 6;
  int s = r % S_DIM;
  int r2 = r / S_DIM;
  int h = r2 % H_NUM;
  int bi = r2 / H_NUM;
  float v = E[i] + (img[0] ? I[i] : 0.0f);
  conc[(((size_t)bi * S_DIM + s) * H_NUM + h) * 64 + dk] = (_Float16)v;
}

// ---------------------------------------------------------------------------
// host orchestration
// d_in order (params dict in insertion order):
//  0..6: q, q_e, q_i, k_e, k_i, v_e, v_i   7: mask   8: image_bool
//  9 q_lin_W 10 q_lin_b 11 q_e_lin_W 12 b 13 q_i_lin_W 14 b 15 k_e_lin_W 16 b
// 17 v_e_lin_W 18 b 19 k_i_lin_W 20 b 21 v_i_lin_W 22 b 23 out_W 24 out_b
// 25 ffn_W1 26 ffn_b1 27 ffn_W2 28 ffn_b2
// ---------------------------------------------------------------------------
extern "C" void kernel_launch(void* const* d_in, const int* in_sizes, int n_in,
                              void* d_out, int out_size, void* d_ws, size_t ws_size,
                              hipStream_t stream) {
  (void)n_in; (void)out_size; (void)ws_size;
  const int B = in_sizes[0] / (S_DIM * D_DIM);
  const int BS = B * S_DIM;
  const size_t BSD = (size_t)B * S_DIM * D_DIM;
  const size_t D2 = (size_t)D_DIM * D_DIM;

  char* ws = (char*)d_ws;
  size_t off = 0;
  auto take = [&](size_t bytes) -> char* {
    char* p = ws + off;
    off = (off + bytes + 255) & ~(size_t)255;
    return p;
  };

  _Float16* x16[7];
  for (int i = 0; i < 7; ++i) x16[i] = (_Float16*)take(BSD * 2);
  _Float16* w16[7];  // q_lin, q_e_lin, q_i_lin, k_e_lin, v_e_lin, k_i_lin, v_i_lin
  for (int i = 0; i < 7; ++i) w16[i] = (_Float16*)take(D2 * 2);
  _Float16* wout16 = (_Float16*)take(D2 * 2);
  _Float16* wf1 = (_Float16*)take((size_t)DK_DIM * DK_DIM * 2);
  _Float16* wf2 = (_Float16*)take((size_t)DK_DIM * DK_DIM * 2);
  _Float16* qh  = (_Float16*)take(BSD * 2);
  _Float16* qeh = (_Float16*)take(BSD * 2);
  _Float16* qih = (_Float16*)take(BSD * 2);
  _Float16* keh = (_Float16*)take(BSD * 2);
  _Float16* kih = (_Float16*)take(BSD * 2);
  _Float16* veh = (_Float16*)take(BSD * 2);
  _Float16* vih = (_Float16*)take(BSD * 2);
  _Float16* tmpA = (_Float16*)take(BSD * 2);
  _Float16* tmpH = (_Float16*)take(BSD * 2);
  _Float16* kei  = (_Float16*)take(BSD * 2);
  _Float16* kie  = (_Float16*)take(BSD * 2);
  float* Ebuf = (float*)take(BSD * 4);
  float* Ibuf = (float*)take(BSD * 4);
  _Float16* conc = (_Float16*)take(BSD * 2);

  const int* maskp = (const int*)d_in[7];
  const int* imgp = (const int*)d_in[8];

  dim3 blk(256);
  int nBSD = (int)BSD;

  auto cvt = [&](const void* src, _Float16* dst, int n) {
    cvt_f32_f16_k<<<dim3((n / 8 + 255) / 256), blk, 0, stream>>>((const float*)src, dst, n);
  };

  // --- convert inputs & weights to f16 ---
  for (int i = 0; i < 7; ++i) cvt(d_in[i], x16[i], nBSD);
  const int wIdx[7] = {9, 11, 13, 15, 17, 19, 21};
  for (int i = 0; i < 7; ++i) cvt(d_in[wIdx[i]], w16[i], (int)D2);
  cvt(d_in[23], wout16, (int)D2);
  cvt(d_in[25], wf1, DK_DIM * DK_DIM);
  cvt(d_in[27], wf2, DK_DIM * DK_DIM);

  auto gemm = [&](const _Float16* A, const _Float16* Bw, const float* bias, void* Out,
                  int M, int N, int K, int mode, int relu) {
    gemm_wmma_k<<<dim3(M / 128, N / 64), blk, 0, stream>>>(A, Bw, bias, Out, M, N, K, mode, relu);
  };
  auto attn = [&](const _Float16* Q, const _Float16* K_, const _Float16* V_,
                  const int* mk, void* Out, int f32, int init) {
    attn_wmma_k<<<dim3(S_DIM / 128, B * H_NUM), blk, 0, stream>>>(Q, K_, V_, mk, Out, f32, init);
  };

  // --- projections into head layout ---
  gemm(x16[0], w16[0], (const float*)d_in[10], qh,  BS, D_DIM, D_DIM, ST_HEAD_F16, 0);
  gemm(x16[1], w16[1], (const float*)d_in[12], qeh, BS, D_DIM, D_DIM, ST_HEAD_F16, 0);
  gemm(x16[2], w16[2], (const float*)d_in[14], qih, BS, D_DIM, D_DIM, ST_HEAD_F16, 0);
  gemm(x16[3], w16[3], (const float*)d_in[16], keh, BS, D_DIM, D_DIM, ST_HEAD_F16, 0);
  gemm(x16[5], w16[4], (const float*)d_in[18], veh, BS, D_DIM, D_DIM, ST_HEAD_F16, 0);
  gemm(x16[4], w16[5], (const float*)d_in[20], kih, BS, D_DIM, D_DIM, ST_HEAD_F16, 0);
  gemm(x16[6], w16[6], (const float*)d_in[22], vih, BS, D_DIM, D_DIM, ST_HEAD_F16, 0);

  const int R = B * H_NUM * S_DIM;  // rows for per-head FFN GEMMs

  // scores_e -> E (init), scores_i -> I (init)
  attn(qh, keh, veh, maskp, Ebuf, 1, 1);
  attn(qh, kih, vih, maskp, Ibuf, 1, 1);

  // kei = ffn(attn(qeh, kih, vih));  scores_ei -> I (+=)
  attn(qeh, kih, vih, nullptr, tmpA, 0, 0);
  gemm(tmpA, wf1, (const float*)d_in[26], tmpH, R, DK_DIM, DK_DIM, ST_ROW_F16, 1);
  gemm(tmpH, wf2, (const float*)d_in[28], kei,  R, DK_DIM, DK_DIM, ST_ROW_F16, 0);
  attn(qh, kei, kei, maskp, Ibuf, 1, 0);

  // kie = ffn(attn(qih, keh, veh));  scores_ie -> I (+=)
  attn(qih, keh, veh, nullptr, tmpA, 0, 0);
  gemm(tmpA, wf1, (const float*)d_in[26], tmpH, R, DK_DIM, DK_DIM, ST_ROW_F16, 1);
  gemm(tmpH, wf2, (const float*)d_in[28], kie,  R, DK_DIM, DK_DIM, ST_ROW_F16, 0);
  attn(qh, kie, kie, maskp, Ibuf, 1, 0);

  // concat (device reads image_bool) and final projection to f32 output
  combine_k<<<dim3((nBSD + 255) / 256), blk, 0, stream>>>(Ebuf, Ibuf, imgp, conc, nBSD);
  gemm(conc, wout16, (const float*)d_in[24], d_out, BS, D_DIM, D_DIM, ST_ROW_F32, 0);
}
